// GCNCombiner_69612829933699
// MI455X (gfx1250) — matrix operand order repaired and checked by reading, not compiled, not run on hardware
//
#include <hip/hip_runtime.h>
#include <hip/hip_bf16.h>
#include <math.h>

// GCNCombiner for MI455X (gfx1250, wave32, WMMA).
// Dominant GEMMs run as v_wmma_f32_16x16x32_f16 with f32 accumulation;
// operands are converted f32->f16 while staging tiles through LDS.
// Staging is vectorized (global b128 loads -> LDS b64 stores); fragment
// reads are contiguous per lane and lower to ds_load_2addr_b64 (verified).

#define B_  32
#define PS_ 1536
#define TNS_ 2048
#define NJ_ 128
#define QK_ 384
#define NC_ 200

#define LAS 36   // lA row stride in halfs (32 + 4 pad; 72 B, 8B-aligned rows)
#define LBS 36   // lBt row stride in halfs

typedef __attribute__((ext_vector_type(16))) _Float16 v16h;
typedef __attribute__((ext_vector_type(8)))  _Float16 v8h;
typedef __attribute__((ext_vector_type(4)))  _Float16 v4h;
typedef __attribute__((ext_vector_type(4)))  float    v4f;
typedef __attribute__((ext_vector_type(8)))  float    v8f;

// Fragment layouts used here (this code's contiguous k order, not the ISA table order):
//   A  : lanes 0-15 row M=lane&15, lanes 16-31 same rows; elem e -> K = e + 16*(lane>>4)
//        => one contiguous 16-half run per lane
//   B  : lane -> col N=lane&15; elem e -> K = e + 16*(lane>>4)
//        => with n-major LDS tile, one contiguous 16-half run
//   C/D: VGPR v -> row M = v + 8*(lane>>4), col N = lane&15
__device__ __forceinline__ v16h load_a(const _Float16* lA, int row, int hiL) {
  v16h a;
#pragma unroll
  for (int e = 0; e < 16; ++e)
    a[e] = lA[row * LAS + e + 16 * hiL];
  return a;
}

__device__ __forceinline__ v16h load_b(const _Float16* lBt, int n, int hiL) {
  v16h b;
#pragma unroll
  for (int e = 0; e < 16; ++e)
    b[e] = lBt[n * LBS + e + 16 * hiL];
  return b;
}

__device__ __forceinline__ v4h cvt4(v4f f) {
  v4h h;
  h[0] = (_Float16)f[0]; h[1] = (_Float16)f[1];
  h[2] = (_Float16)f[2]; h[3] = (_Float16)f[3];
  return h;
}

// ---- Step A: hs0[bc, j] = sum_s x[bc, s] * w_pool0[j, s] + b_pool0[j]  (f16 out) ----
// M = B*PS = 49152 (grid.x tiles of 32), N = NJ = 128 (8 waves x 16), K = TNS = 2048.
// Each wave owns two 16x16 output tiles (M-halves) sharing one B fragment.
__global__ __launch_bounds__(256) void k_gemm_pool0(const float* __restrict__ x,
                                                    const float* __restrict__ w0,
                                                    const float* __restrict__ b0,
                                                    _Float16* __restrict__ hs0) {
  __shared__ _Float16 lA[32 * LAS];
  __shared__ _Float16 lB[128 * LBS];   // n-major (transposed) B tile
  const int tid  = threadIdx.x;
  const int lane = tid & 31, wave = tid >> 5;
  const int row0 = blockIdx.x * 32;
  const int col = lane & 15, hiL = lane >> 4, n0 = wave * 16;
  // A staging slot: 32x32 tile, 4 halfs per thread
  const int ar = tid >> 3, ak = (tid & 7) * 4;
  v8f acc0 = {}, acc1 = {};
  for (int kc = 0; kc < TNS_; kc += 32) {
    __syncthreads();
    // A tile: one b128 f32 load + one b64 LDS store per thread
    v4f av = *(const v4f*)&x[(size_t)(row0 + ar) * TNS_ + kc + ak];
    *(v4h*)&lA[ar * LAS + ak] = cvt4(av);
    // B tile: lBt[n][k] = w0[n*TNS + k]; 4x (b128 load + b64 store) per thread
#pragma unroll
    for (int i = 0; i < 4; ++i) {
      int idx = (i * 256 + tid) * 4;           // 0..4095
      int n = idx >> 5, kk = idx & 31;
      v4f wv = *(const v4f*)&w0[(size_t)n * TNS_ + kc + kk];
      *(v4h*)&lB[n * LBS + kk] = cvt4(wv);
    }
    __syncthreads();
    v16h bf = load_b(lB, n0 + col, hiL);
    v16h a0 = load_a(lA, col, hiL);
    v16h a1 = load_a(lA, 16 + col, hiL);
    acc0 = __builtin_amdgcn_wmma_f32_16x16x32_f16(false, a0, false, bf, (short)0, acc0, false, false);
    acc1 = __builtin_amdgcn_wmma_f32_16x16x32_f16(false, a1, false, bf, (short)0, acc1, false, false);
  }
  const float bias = b0[n0 + col];
#pragma unroll
  for (int v = 0; v < 8; ++v) {
    int m = v + 8 * hiL;
    hs0[(size_t)(row0 + m) * NJ_ + n0 + col]      = (_Float16)(acc0[v] + bias);
    hs0[(size_t)(row0 + 16 + m) * NJ_ + n0 + col] = (_Float16)(acc1[v] + bias);
  }
}

// ---- Step B: hs2[b, o, j] = sum_c w_c1[o, c] * hs0[b, c, j] + b_c1[o]  (f16 out) ----
__global__ __launch_bounds__(256) void k_gemm_conv1(const float* __restrict__ wc1,
                                                    const float* __restrict__ bc1,
                                                    const _Float16* __restrict__ hs0,
                                                    _Float16* __restrict__ hs2) {
  __shared__ _Float16 lA[32 * LAS];
  __shared__ _Float16 lB[128 * LBS];
  const int tid  = threadIdx.x;
  const int lane = tid & 31, wave = tid >> 5;
  const int m0 = blockIdx.x * 32;              // over PS (output channels o)
  const int bb = blockIdx.y;                   // batch
  const int col = lane & 15, hiL = lane >> 4, n0 = wave * 16;
  const int ar = tid >> 3, ak = (tid & 7) * 4;
  const _Float16* hsB = hs0 + (size_t)bb * PS_ * NJ_;
  v8f acc0 = {}, acc1 = {};
  for (int kc = 0; kc < PS_; kc += 32) {
    __syncthreads();
    // A tile: 32x32 from w_c1 (f32 -> f16), vectorized
    v4f av = *(const v4f*)&wc1[(size_t)(m0 + ar) * PS_ + kc + ak];
    *(v4h*)&lA[ar * LAS + ak] = cvt4(av);
    // B tile: lBt[n][k] = hs0[b, kc+k, n]; b128 coalesced loads, transposed stores
#pragma unroll
    for (int i = 0; i < 2; ++i) {
      int idx = (i * 256 + tid) * 8;           // 0..4095
      int kk = idx >> 7, n = idx & 127;        // n multiple of 8
      v8h hv = *(const v8h*)&hsB[(size_t)(kc + kk) * NJ_ + n];
#pragma unroll
      for (int e = 0; e < 8; ++e) lB[(n + e) * LBS + kk] = hv[e];
    }
    __syncthreads();
    v16h bf = load_b(lB, n0 + col, hiL);
    v16h a0 = load_a(lA, col, hiL);
    v16h a1 = load_a(lA, 16 + col, hiL);
    acc0 = __builtin_amdgcn_wmma_f32_16x16x32_f16(false, a0, false, bf, (short)0, acc0, false, false);
    acc1 = __builtin_amdgcn_wmma_f32_16x16x32_f16(false, a1, false, bf, (short)0, acc1, false, false);
  }
#pragma unroll
  for (int v = 0; v < 8; ++v) {
    int m = v + 8 * hiL;
    hs2[((size_t)bb * PS_ + m0 + m) * NJ_ + n0 + col] =
        (_Float16)(acc0[v] + bc1[m0 + m]);
    hs2[((size_t)bb * PS_ + m0 + 16 + m) * NJ_ + n0 + col] =
        (_Float16)(acc1[v] + bc1[m0 + 16 + m]);
  }
}

// ---- Step C: hs3[b, c, k] = sum_j hs2[b, c, j] * A1[b, j, k]  (f32 out for BN) ----
__global__ __launch_bounds__(256) void k_gemm_adj(const _Float16* __restrict__ hs2,
                                                  const _Float16* __restrict__ A1,
                                                  float* __restrict__ hs3) {
  __shared__ _Float16 lA[32 * LAS];
  __shared__ _Float16 lB[128 * LBS];
  const int tid  = threadIdx.x;
  const int lane = tid & 31, wave = tid >> 5;
  const int m0 = blockIdx.x * 32;              // over PS (channels c)
  const int bb = blockIdx.y;                   // batch
  const int col = lane & 15, hiL = lane >> 4, n0 = wave * 16;
  const int ar = tid >> 3, ak = (tid & 7) * 4;
  const _Float16* A1b = A1 + (size_t)bb * NJ_ * NJ_;
  v8f acc0 = {}, acc1 = {};
  for (int kc = 0; kc < NJ_; kc += 32) {
    __syncthreads();
    // A tile: hs2[b, m0+r, kc+kk] (f16 source), b64 load + b64 store
    *(v4h*)&lA[ar * LAS + ak] =
        *(const v4h*)&hs2[((size_t)bb * PS_ + m0 + ar) * NJ_ + kc + ak];
    // B tile: lBt[n][k] = A1[b, kc+k, n]
#pragma unroll
    for (int i = 0; i < 2; ++i) {
      int idx = (i * 256 + tid) * 8;
      int kk = idx >> 7, n = idx & 127;
      v8h hv = *(const v8h*)&A1b[(size_t)(kc + kk) * NJ_ + n];
#pragma unroll
      for (int e = 0; e < 8; ++e) lB[(n + e) * LBS + kk] = hv[e];
    }
    __syncthreads();
    v16h bf = load_b(lB, n0 + col, hiL);
    v16h a0 = load_a(lA, col, hiL);
    v16h a1 = load_a(lA, 16 + col, hiL);
    acc0 = __builtin_amdgcn_wmma_f32_16x16x32_f16(false, a0, false, bf, (short)0, acc0, false, false);
    acc1 = __builtin_amdgcn_wmma_f32_16x16x32_f16(false, a1, false, bf, (short)0, acc1, false, false);
  }
#pragma unroll
  for (int v = 0; v < 8; ++v) {
    int m = v + 8 * hiL;
    hs3[((size_t)bb * PS_ + m0 + m) * NJ_ + n0 + col]      = acc0[v];
    hs3[((size_t)bb * PS_ + m0 + 16 + m) * NJ_ + n0 + col] = acc1[v];
  }
}

// ---- fold the q/k 1x1 convs + channel-mean into per-channel weight means ----
__global__ void k_wqk_mean(const float* __restrict__ wq, const float* __restrict__ bq,
                           const float* __restrict__ wk, const float* __restrict__ bk,
                           float* __restrict__ wqm, float* __restrict__ wkm,
                           float* __restrict__ bqkm) {
  int c = blockIdx.x * blockDim.x + threadIdx.x;
  if (c < PS_) {
    float sq = 0.f, sk = 0.f;
    for (int o = 0; o < QK_; ++o) {
      sq += wq[(size_t)o * PS_ + c];
      sk += wk[(size_t)o * PS_ + c];
    }
    wqm[c] = sq / (float)QK_;
    wkm[c] = sk / (float)QK_;
  }
  if (blockIdx.x == 0 && threadIdx.x == 0) {
    float sbq = 0.f, sbk = 0.f;
    for (int o = 0; o < QK_; ++o) { sbq += bq[o]; sbk += bk[o]; }
    bqkm[0] = sbq / (float)QK_;
    bqkm[1] = sbk / (float)QK_;
  }
}

// ---- q1[b,j], k1[b,j] from hs0 (f16) and folded weight means ----
__global__ void k_qk(const _Float16* __restrict__ hs0, const float* __restrict__ wqm,
                     const float* __restrict__ wkm, const float* __restrict__ bqkm,
                     float* __restrict__ q1, float* __restrict__ k1) {
  int bb = blockIdx.x;
  int j = threadIdx.x;  // 128 threads
  float sq = 0.f, sk = 0.f;
  for (int c = 0; c < PS_; ++c) {
    float h = (float)hs0[((size_t)bb * PS_ + c) * NJ_ + j];
    sq += wqm[c] * h;
    sk += wkm[c] * h;
  }
  q1[bb * NJ_ + j] = sq + bqkm[0];
  k1[bb * NJ_ + j] = sk + bqkm[1];
}

// ---- A1[b,j,k] = adj1[j,k] + tanh(q1[b,j]-k1[b,k])*alpha  (f16 out) ----
__global__ void k_A1(const float* __restrict__ adj1, const float* __restrict__ q1,
                     const float* __restrict__ k1, const float* __restrict__ alpha,
                     _Float16* __restrict__ A1) {
  int idx = blockIdx.x * blockDim.x + threadIdx.x;  // 32*128*128
  int bb = idx >> 14;
  int rem = idx & 16383;
  int j = rem >> 7, kk = rem & 127;
  float v = adj1[j * NJ_ + kk] +
            tanhf(q1[bb * NJ_ + j] - k1[bb * NJ_ + kk]) * alpha[0];
  A1[idx] = (_Float16)v;
}

// ---- BN stats per channel c over (b, j): 4096 values ----
__global__ __launch_bounds__(256) void k_bnstats(const float* __restrict__ hs3,
                                                 float* __restrict__ meanv,
                                                 float* __restrict__ rstdv) {
  __shared__ float ssum[256], ssq[256];
  int c = blockIdx.x, tid = threadIdx.x;
  float s = 0.f, q = 0.f;
  for (int t = tid; t < B_ * NJ_; t += 256) {
    int bb = t >> 7, j = t & 127;
    float v = hs3[((size_t)bb * PS_ + c) * NJ_ + j];
    s += v; q += v * v;
  }
  ssum[tid] = s; ssq[tid] = q;
  __syncthreads();
  for (int st = 128; st > 0; st >>= 1) {
    if (tid < st) { ssum[tid] += ssum[tid + st]; ssq[tid] += ssq[tid + st]; }
    __syncthreads();
  }
  if (tid == 0) {
    float m = ssum[0] / (float)(B_ * NJ_);
    float var = ssq[0] / (float)(B_ * NJ_) - m * m;
    meanv[c] = m;
    rstdv[c] = rsqrtf(var + 1e-5f);
  }
}

// ---- BN apply + param_pool1 (NJ -> 1): hs4[b,c] ----
__global__ void k_pool1(const float* __restrict__ hs3, const float* __restrict__ meanv,
                        const float* __restrict__ rstdv, const float* __restrict__ gamma,
                        const float* __restrict__ beta, const float* __restrict__ wp1,
                        const float* __restrict__ bp1, float* __restrict__ hs4) {
  int id = blockIdx.x * blockDim.x + threadIdx.x;  // 32*1536
  int bb = id / PS_, c = id % PS_;
  float m = meanv[c], rs = rstdv[c], g = gamma[c], bt = beta[c];
  const float* row = hs3 + ((size_t)bb * PS_ + c) * NJ_;
  float s = 0.f;
  for (int j = 0; j < NJ_; ++j) {
    float v = g * (row[j] - m) * rs + bt;
    s += v * wp1[j];
  }
  hs4[id] = s + bp1[0];
}

// ---- classifier: out[b,n] = hs4[b,:] . w_cls[n,:] + b_cls[n] ----
__global__ void k_cls(const float* __restrict__ hs4, const float* __restrict__ wcls,
                      const float* __restrict__ bcls, float* __restrict__ out) {
  int id = blockIdx.x * blockDim.x + threadIdx.x;  // 32*200
  if (id >= B_ * NC_) return;
  int bb = id / NC_, n = id % NC_;
  const float* hv = hs4 + (size_t)bb * PS_;
  const float* wv = wcls + (size_t)n * PS_;
  float s = 0.f;
  for (int c = 0; c < PS_; ++c) s += hv[c] * wv[c];
  out[id] = s + bcls[n];
}

extern "C" void kernel_launch(void* const* d_in, const int* in_sizes, int n_in,
                              void* d_out, int out_size, void* d_ws, size_t ws_size,
                              hipStream_t stream) {
  const float* x       = (const float*)d_in[0];
  const float* w_pool0 = (const float*)d_in[1];
  const float* b_pool0 = (const float*)d_in[2];
  const float* adj1    = (const float*)d_in[3];
  const float* w_q     = (const float*)d_in[4];
  const float* b_q     = (const float*)d_in[5];
  const float* w_k     = (const float*)d_in[6];
  const float* b_k     = (const float*)d_in[7];
  const float* alpha   = (const float*)d_in[8];
  const float* w_c1    = (const float*)d_in[9];
  const float* b_c1    = (const float*)d_in[10];
  const float* gamma   = (const float*)d_in[11];
  const float* beta    = (const float*)d_in[12];
  const float* w_pool1 = (const float*)d_in[13];
  const float* b_pool1 = (const float*)d_in[14];
  const float* w_cls   = (const float*)d_in[15];
  const float* b_cls   = (const float*)d_in[16];
  float* out = (float*)d_out;

  char* ws = (char*)d_ws;
  size_t off = 0;
  auto salloc = [&](size_t bytes) -> void* {
    void* p = ws + off;
    off = (off + bytes + 255) & ~(size_t)255;
    return p;
  };
  const size_t NHS = (size_t)B_ * PS_ * NJ_;        // 6,291,456
  _Float16* hs0  = (_Float16*)salloc(NHS * 2);
  _Float16* hs2  = (_Float16*)salloc(NHS * 2);
  _Float16* A1f  = (_Float16*)salloc((size_t)B_ * NJ_ * NJ_ * 2);
  float*    hs3  = (float*)salloc(NHS * 4);
  float*    q1   = (float*)salloc((size_t)B_ * NJ_ * 4);
  float*    k1   = (float*)salloc((size_t)B_ * NJ_ * 4);
  float*    wqm  = (float*)salloc(PS_ * 4);
  float*    wkm  = (float*)salloc(PS_ * 4);
  float*    bqkm = (float*)salloc(2 * 4);
  float*    meanv= (float*)salloc(PS_ * 4);
  float*    rstdv= (float*)salloc(PS_ * 4);
  float*    hs4  = (float*)salloc((size_t)B_ * PS_ * 4);

  // 1) fold q/k weights
  k_wqk_mean<<<(PS_ + 255) / 256, 256, 0, stream>>>(w_q, b_q, w_k, b_k, wqm, wkm, bqkm);
  // 2) pool0 GEMM (dominant, WMMA): M=B*PS, N=NJ, K=TNS
  k_gemm_pool0<<<(B_ * PS_) / 32, 256, 0, stream>>>(x, w_pool0, b_pool0, hs0);
  // 3) q1/k1
  k_qk<<<B_, NJ_, 0, stream>>>(hs0, wqm, wkm, bqkm, q1, k1);
  // 4) attention-blended adjacency
  k_A1<<<(B_ * NJ_ * NJ_) / 256, 256, 0, stream>>>(adj1, q1, k1, alpha, A1f);
  // 5) conv1 GEMM (WMMA): per-batch M=PS, N=NJ, K=PS
  k_gemm_conv1<<<dim3(PS_ / 32, B_), 256, 0, stream>>>(w_c1, b_c1, hs0, hs2);
  // 6) adjacency GEMM (WMMA): per-batch M=PS, N=NJ, K=NJ
  k_gemm_adj<<<dim3(PS_ / 32, B_), 256, 0, stream>>>(hs2, A1f, hs3);
  // 7) batch-norm stats
  k_bnstats<<<PS_, 256, 0, stream>>>(hs3, meanv, rstdv);
  // 8) BN apply + pool1
  k_pool1<<<(B_ * PS_) / 256, 256, 0, stream>>>(hs3, meanv, rstdv, gamma, beta,
                                                w_pool1, b_pool1, hs4);
  // 9) classifier
  k_cls<<<(B_ * NC_ + 255) / 256, 256, 0, stream>>>(hs4, w_cls, b_cls, out);
}
